// NESCL_52458730553686
// MI455X (gfx1250) — matrix operand order repaired and checked by reading, not compiled, not run on hardware
//
#include <hip/hip_runtime.h>
#include <hip/hip_bf16.h>
#include <math.h>

typedef __attribute__((ext_vector_type(16))) _Float16 v16h;
typedef __attribute__((ext_vector_type(8)))  _Float16 v8h;
typedef __attribute__((ext_vector_type(8)))  float    v8f;

#define N_USERS 60000
#define N_ITEMS 40000
#define DIM     64
#define SSL_TEMP 0.1f
#define SSL_REG  0.3f
#define ITEM_TILES (N_ITEMS / 16)   // 2500
#define LSE_SHIFT 10.0f             // |logit| <= 1/SSL_TEMP = 10 (unit vectors)
// exp(logit - 10) == exp2( fma(dot, K2, -K2) ), K2 = 10*log2(e)
#define K2_LOG2E 14.4269504088896340736f

// ---------------- zero prefix (float4) ----------------
__global__ void k_zero4(float4* __restrict__ p, int n4) {
    int i = blockIdx.x * blockDim.x + threadIdx.x;
    if (i < n4) p[i] = make_float4(0.f, 0.f, 0.f, 0.f);
}

// ---------------- degrees ----------------
__global__ void k_degree(const int* __restrict__ eu, const int* __restrict__ ei,
                         float* __restrict__ du, float* __restrict__ di, int E) {
    int e = blockIdx.x * blockDim.x + threadIdx.x;
    if (e >= E) return;
    atomicAdd(&du[eu[e]], 1.0f);
    atomicAdd(&di[ei[e]], 1.0f);
}

// ---------------- edge weights: rsqrt(deg_u * deg_i) ----------------
__global__ void k_weight(const int* __restrict__ eu, const int* __restrict__ ei,
                         const float* __restrict__ du, const float* __restrict__ di,
                         float* __restrict__ w, int E) {
    int e = blockIdx.x * blockDim.x + threadIdx.x;
    if (e >= E) return;
    w[e] = rsqrtf(fmaxf(du[eu[e]], 1.0f) * fmaxf(di[ei[e]], 1.0f));
}

// ---------------- one propagation layer, both directions, wave-per-edge ----------------
__global__ void k_scatter(const float* __restrict__ src_u, const float* __restrict__ src_i,
                          float* __restrict__ dst_u, float* __restrict__ dst_i,
                          const int* __restrict__ eu, const int* __restrict__ ei,
                          const float* __restrict__ w, int E) {
    int gid  = blockIdx.x * blockDim.x + threadIdx.x;
    int e    = gid >> 5;
    int lane = gid & 31;
    if (e >= E) return;
    int u = eu[e], it = ei[e];
    float ww = w[e];
    int d = lane * 2;
    float2 iv = *reinterpret_cast<const float2*>(src_i + (size_t)it * DIM + d);
    float2 uv = *reinterpret_cast<const float2*>(src_u + (size_t)u  * DIM + d);
    float* pu = dst_u + (size_t)u  * DIM + d;
    float* pi = dst_i + (size_t)it * DIM + d;
    atomicAdd(pu,     ww * iv.x);
    atomicAdd(pu + 1, ww * iv.y);
    atomicAdd(pi,     ww * uv.x);
    atomicAdd(pi + 1, ww * uv.y);
}

// ---------------- i_final = e0+m1+m2 (scale cancels under L2-norm) -> z_all (f16) ----------------
__global__ void k_item_norm(const float* __restrict__ emb, const float* __restrict__ m1,
                            const float* __restrict__ m2, _Float16* __restrict__ zall, int n) {
    int gid  = blockIdx.x * blockDim.x + threadIdx.x;
    int row  = gid >> 5;
    int lane = gid & 31;
    if (row >= n) return;
    size_t off = (size_t)row * DIM + lane * 2;
    float2 a = *reinterpret_cast<const float2*>(emb + off);
    float2 b = *reinterpret_cast<const float2*>(m1  + off);
    float2 c = *reinterpret_cast<const float2*>(m2  + off);
    float x = a.x + b.x + c.x, y = a.y + b.y + c.y;
    float ss = x * x + y * y;
    #pragma unroll
    for (int o = 16; o; o >>= 1) ss += __shfl_xor(ss, o, 32);
    float inv = rsqrtf(fmaxf(ss, 1e-24f));
    zall[off]     = (_Float16)(x * inv);
    zall[off + 1] = (_Float16)(y * inv);
}

// ---------------- gather batch users/items, write zu (f16), pos[b] ----------------
__global__ void k_user_gather(const float* __restrict__ uemb, const float* __restrict__ um1,
                              const float* __restrict__ um2,
                              const float* __restrict__ iemb, const float* __restrict__ im1,
                              const float* __restrict__ im2,
                              const int* __restrict__ busers, const int* __restrict__ bitems,
                              _Float16* __restrict__ zu, float* __restrict__ pos, int B) {
    int gid  = blockIdx.x * blockDim.x + threadIdx.x;
    int b    = gid >> 5;
    int lane = gid & 31;
    if (b >= B) return;
    int ur = busers[b], ir = bitems[b];
    size_t uo = (size_t)ur * DIM + lane * 2;
    size_t io = (size_t)ir * DIM + lane * 2;
    float2 a0 = *reinterpret_cast<const float2*>(uemb + uo);
    float2 a1 = *reinterpret_cast<const float2*>(um1  + uo);
    float2 a2 = *reinterpret_cast<const float2*>(um2  + uo);
    float2 c0 = *reinterpret_cast<const float2*>(iemb + io);
    float2 c1 = *reinterpret_cast<const float2*>(im1  + io);
    float2 c2 = *reinterpret_cast<const float2*>(im2  + io);
    float ux = a0.x + a1.x + a2.x, uy = a0.y + a1.y + a2.y;
    float ix = c0.x + c1.x + c2.x, iy = c0.y + c1.y + c2.y;
    float su = ux * ux + uy * uy;
    float si = ix * ix + iy * iy;
    float dp = ux * ix + uy * iy;
    #pragma unroll
    for (int o = 16; o; o >>= 1) {
        su += __shfl_xor(su, o, 32);
        si += __shfl_xor(si, o, 32);
        dp += __shfl_xor(dp, o, 32);
    }
    float invu = rsqrtf(fmaxf(su, 1e-24f));
    float invi = rsqrtf(fmaxf(si, 1e-24f));
    size_t zo = (size_t)b * DIM + lane * 2;
    zu[zo]     = (_Float16)(ux * invu);
    zu[zo + 1] = (_Float16)(uy * invu);
    if (lane == 0) pos[b] = dp * invu * invi * (1.0f / SSL_TEMP);
}

// ---------------- WMMA logits + fixed-shift streaming logsumexp ----------------
// |logit| <= 10 (unit vectors) -> exact logsumexp with fixed shift 10:
//   S += exp2(fma(dot, K2, -K2)),  lse = 10 + log(S)
// Two column tiles per iteration: c0's readout is separated from its WMMAs
// by c1's WMMAs + the prefetch clause (fills the f16 WMMA->VALU hazard), and
// the single log2-domain FMA feeds v_exp_f32 directly.
__global__ void k_logits(const _Float16* __restrict__ zu, const _Float16* __restrict__ zall,
                         const float* __restrict__ pos, float* __restrict__ acc) {
    __shared__ float lds_s[8][16];
    const int wv   = __builtin_amdgcn_readfirstlane(threadIdx.x >> 5);  // uniform wave id
    const int lane = threadIdx.x & 31;
    const int m  = lane & 15;
    const int hk = lane >> 4;

    // A fragments (contiguous K, matching B): lanes 0-15 -> K 0..15,
    // lanes 16-31 -> K 16..31; second WMMA covers K base +32.
    const _Float16* arow = zu + (size_t)(blockIdx.x * 16 + m) * DIM;
    union AU { v16h v; v8h h[2]; };
    AU a0u, a1u;
    a0u.h[0] = *reinterpret_cast<const v8h*>(arow + hk * 16);
    a0u.h[1] = *reinterpret_cast<const v8h*>(arow + 8 + hk * 16);
    a1u.h[0] = *reinterpret_cast<const v8h*>(arow + 32 + hk * 16);
    a1u.h[1] = *reinterpret_cast<const v8h*>(arow + 40 + hk * 16);
    const v16h a0 = a0u.v, a1 = a1u.v;

    float srun[8];
    #pragma unroll
    for (int v = 0; v < 8; ++v) srun[v] = 0.0f;

    // per-lane B base: column (lane&15) of each tile, K slice per lane half
    const _Float16* bbase = zall + (size_t)(lane & 15) * DIM + hk * 16;
    const size_t tile_halves = (size_t)16 * DIM;   // halves per 16-item tile

    int ct = wv;                                   // pair = (ct, ct+8)
    const _Float16* bp0 = bbase + (size_t)ct * tile_halves;
    const _Float16* bp1 = bbase + (size_t)(ct + 8) * tile_halves;
    v16h b0 = *reinterpret_cast<const v16h*>(bp0);
    v16h b1 = *reinterpret_cast<const v16h*>(bp0 + 32);
    v16h b2 = *reinterpret_cast<const v16h*>(bp1);
    v16h b3 = *reinterpret_cast<const v16h*>(bp1 + 32);

    while (ct + 8 < ITEM_TILES) {
        v8f c0 = {}, c1 = {};
        c0 = __builtin_amdgcn_wmma_f32_16x16x32_f16(false, a0, false, b0, (short)0, c0, false, false);
        c0 = __builtin_amdgcn_wmma_f32_16x16x32_f16(false, a1, false, b1, (short)0, c0, false, false);
        c1 = __builtin_amdgcn_wmma_f32_16x16x32_f16(false, a0, false, b2, (short)0, c1, false, false);
        c1 = __builtin_amdgcn_wmma_f32_16x16x32_f16(false, a1, false, b3, (short)0, c1, false, false);
        // branchless prefetch of next pair (clamped; redundant on last iteration)
        int t0 = ct + 16, t1 = ct + 24;
        int l0 = (t0 < ITEM_TILES) ? t0 : ct;
        int l1 = (t1 < ITEM_TILES) ? t1 : ct;
        const _Float16* q0 = bbase + (size_t)l0 * tile_halves;
        const _Float16* q1 = bbase + (size_t)l1 * tile_halves;
        b0 = *reinterpret_cast<const v16h*>(q0);
        b1 = *reinterpret_cast<const v16h*>(q0 + 32);
        b2 = *reinterpret_cast<const v16h*>(q1);
        b3 = *reinterpret_cast<const v16h*>(q1 + 32);
        #pragma unroll
        for (int v = 0; v < 8; ++v)
            srun[v] += __builtin_amdgcn_exp2f(fmaf(c0[v], K2_LOG2E, -K2_LOG2E));
        #pragma unroll
        for (int v = 0; v < 8; ++v)
            srun[v] += __builtin_amdgcn_exp2f(fmaf(c1[v], K2_LOG2E, -K2_LOG2E));
        ct = t0;
    }
    // leftover single tile (waves 0-3: tiles 2496..2499); b0/b1 already hold it
    if (ct < ITEM_TILES) {
        v8f c0 = {};
        c0 = __builtin_amdgcn_wmma_f32_16x16x32_f16(false, a0, false, b0, (short)0, c0, false, false);
        c0 = __builtin_amdgcn_wmma_f32_16x16x32_f16(false, a1, false, b1, (short)0, c0, false, false);
        #pragma unroll
        for (int v = 0; v < 8; ++v)
            srun[v] += __builtin_amdgcn_exp2f(fmaf(c0[v], K2_LOG2E, -K2_LOG2E));
    }

    // plain-sum butterfly across the 16 lanes of each half (C layout:
    // rows 0-7 in lanes 0-15 at c[v]=row v; rows 8-15 in lanes 16-31).
    #pragma unroll
    for (int o = 8; o; o >>= 1) {
        #pragma unroll
        for (int v = 0; v < 8; ++v) srun[v] += __shfl_xor(srun[v], o, 32);
    }
    if ((lane & 15) == 0) {
        #pragma unroll
        for (int v = 0; v < 8; ++v) lds_s[wv][hk * 8 + v] = srun[v];
    }
    __syncthreads();

    if (threadIdx.x < 16) {
        int r = threadIdx.x;
        float S = 0.0f;
        #pragma unroll
        for (int w8 = 0; w8 < 8; ++w8) S += lds_s[w8][r];
        float lse = LSE_SHIFT + __logf(S);
        int b = blockIdx.x * 16 + r;
        atomicAdd(acc, pos[b] - lse);
    }
}

__global__ void k_finalize(const float* __restrict__ acc, float* __restrict__ out, float scale) {
    out[0] = scale * acc[0];
}

extern "C" void kernel_launch(void* const* d_in, const int* in_sizes, int n_in,
                              void* d_out, int out_size, void* d_ws, size_t ws_size,
                              hipStream_t stream) {
    const float* user_emb = (const float*)d_in[0];
    const float* item_emb = (const float*)d_in[1];
    const int*   eu       = (const int*)d_in[2];
    const int*   ei       = (const int*)d_in[3];
    const int*   busers   = (const int*)d_in[4];
    const int*   bitems   = (const int*)d_in[5];
    const int E = in_sizes[2];
    const int B = in_sizes[4];

    char* ws = (char*)d_ws;
    size_t off = 0;
    auto take = [&](size_t bytes) { size_t r = off; off += bytes; return r; };
    // zero-prefix region (all sizes multiples of 16 B; base 256B-aligned by hipMalloc)
    float* deg_u = (float*)(ws + take((size_t)N_USERS * 4));
    float* deg_i = (float*)(ws + take((size_t)N_ITEMS * 4));
    float* u_m1  = (float*)(ws + take((size_t)N_USERS * DIM * 4));
    float* i_m1  = (float*)(ws + take((size_t)N_ITEMS * DIM * 4));
    float* u_m2  = (float*)(ws + take((size_t)N_USERS * DIM * 4));
    float* i_m2  = (float*)(ws + take((size_t)N_ITEMS * DIM * 4));
    float* acc   = (float*)(ws + take(256));
    const size_t zero_bytes = off;
    // non-zeroed scratch
    float*     w    = (float*)(ws + take((size_t)E * 4));
    _Float16*  zall = (_Float16*)(ws + take((size_t)N_ITEMS * DIM * 2));
    _Float16*  zu   = (_Float16*)(ws + take((size_t)B * DIM * 2));
    float*     pos  = (float*)(ws + take((size_t)B * 4));
    (void)ws_size; (void)n_in;

    const int T = 256;
    // 1) zero degrees, message buffers, accumulator
    int n4 = (int)(zero_bytes / 16);
    k_zero4<<<(n4 + T - 1) / T, T, 0, stream>>>((float4*)ws, n4);
    // 2) degrees
    k_degree<<<(E + T - 1) / T, T, 0, stream>>>(eu, ei, deg_u, deg_i, E);
    // 3) symmetric normalization weights
    k_weight<<<(E + T - 1) / T, T, 0, stream>>>(eu, ei, deg_u, deg_i, w, E);
    // 4) layer 1: from raw embeddings
    long long thr = (long long)E * 32;
    k_scatter<<<(unsigned)((thr + T - 1) / T), T, 0, stream>>>(user_emb, item_emb, u_m1, i_m1, eu, ei, w, E);
    // 5) layer 2: from layer-1 messages
    k_scatter<<<(unsigned)((thr + T - 1) / T), T, 0, stream>>>(u_m1, i_m1, u_m2, i_m2, eu, ei, w, E);
    // 6) all-item i_final -> normalized f16 z_all
    k_item_norm<<<(N_ITEMS * 32 + T - 1) / T, T, 0, stream>>>(item_emb, i_m1, i_m2, zall, N_ITEMS);
    // 7) batch gather: zu (f16) + pos
    k_user_gather<<<(B * 32 + T - 1) / T, T, 0, stream>>>(user_emb, u_m1, u_m2,
                                                          item_emb, i_m1, i_m2,
                                                          busers, bitems, zu, pos, B);
    // 8) WMMA logits + fixed-shift streaming logsumexp (256 row tiles x 8 waves)
    k_logits<<<B / 16, 256, 0, stream>>>(zu, zall, pos, acc);
    // 9) loss = -SSL_REG * mean(pos - lse)
    k_finalize<<<1, 1, 0, stream>>>(acc, (float*)d_out, -SSL_REG / (float)B);
}
